// GroupAttention2_2851858284545
// MI455X (gfx1250) — matrix operand order repaired and checked by reading, not compiled, not run on hardware
//
#include <hip/hip_runtime.h>
#include <hip/hip_bf16.h>
#include <math.h>

// ---------------------------------------------------------------------------
// Problem constants (match reference: B=16, S_FULL=1024, S=512, D=1024)
// ---------------------------------------------------------------------------
#define Bn    16
#define SFULL 1024
#define Sev   512
#define Dm    1024
#define EPSc  1e-15f
#define LNEPS 1e-12f

typedef __attribute__((ext_vector_type(16))) _Float16 v16h;
typedef __attribute__((ext_vector_type(8)))  float    v8f;
typedef __attribute__((ext_vector_type(4)))  float    v4f;

// POD union for fragment bit-casts (HIP float4 has ctors; v4f is POD-safe)
union Frag16 { v4f f[2]; v16h h; };

// ---------------------------------------------------------------------------
// Kernel 0: convert Wk / Wq (f32, row-major [out, in]) to f16 for WMMA
// ---------------------------------------------------------------------------
__global__ void k_cvt_w(const float* __restrict__ wk, const float* __restrict__ wq,
                        _Float16* __restrict__ wk16, _Float16* __restrict__ wq16) {
  int i = blockIdx.x * blockDim.x + threadIdx.x;
  if (i < Dm * Dm) {
    wk16[i] = (_Float16)wk[i];
    wq16[i] = (_Float16)wq[i];
  }
}

// ---------------------------------------------------------------------------
// Kernel 1: gather rows by ip_event_loc + LayerNorm, emit f16 ctx
// One 256-thread block per (b,s) row of 1024 elements.
// ---------------------------------------------------------------------------
__global__ __launch_bounds__(256)
void k_gather_ln(const float* __restrict__ hs, const int* __restrict__ loc,
                 const float* __restrict__ lnw, const float* __restrict__ lnb,
                 _Float16* __restrict__ ctx) {
  int row = blockIdx.x;                 // b*Sev + s
  int b = row / Sev, s = row % Sev;
  int src_s = loc[b * Sev + s];
  const float* src = hs + ((size_t)b * SFULL + (size_t)src_s) * Dm;
  int t = threadIdx.x;

  float x[4], sum = 0.f, sq = 0.f;
#pragma unroll
  for (int i = 0; i < 4; i++) {
    x[i] = src[t + 256 * i];
    sum += x[i];
    sq  += x[i] * x[i];
  }
  // intra-wave reduce (wave32)
  for (int o = 16; o > 0; o >>= 1) {
    sum += __shfl_xor(sum, o, 32);
    sq  += __shfl_xor(sq,  o, 32);
  }
  __shared__ float ssum[8], ssq[8];
  int w = t >> 5, l = t & 31;
  if (l == 0) { ssum[w] = sum; ssq[w] = sq; }
  __syncthreads();
  if (w == 0) {
    float a = (l < 8) ? ssum[l] : 0.f;
    float c = (l < 8) ? ssq[l]  : 0.f;
    for (int o = 4; o > 0; o >>= 1) {
      a += __shfl_xor(a, o, 32);
      c += __shfl_xor(c, o, 32);
    }
    if (l == 0) { ssum[0] = a; ssq[0] = c; }
  }
  __syncthreads();
  float mu  = ssum[0] * (1.0f / Dm);
  float var = ssq[0] * (1.0f / Dm) - mu * mu;
  float rs  = rsqrtf(var + LNEPS);

  _Float16* dst = ctx + (size_t)row * Dm;
#pragma unroll
  for (int i = 0; i < 4; i++) {
    int d = t + 256 * i;
    float v = (x[i] - mu) * rs * lnw[d] + lnb[d];
    dst[d] = (_Float16)v;
  }
}

// ---------------------------------------------------------------------------
// Kernel 2: WMMA GEMM  out[s,o] = sum_d ctx[s,d] * W[o,d] + bias[o]
// Block = 256 threads (8 waves), computes 64x128 output tile.
// Software-pipelined: double-buffered LDS tiles, global->reg prefetch of the
// next tile issued before the WMMA chain, all fragments preloaded so the 4
// v_wmma ops issue back-to-back after one dscnt wait.
// Fragment layouts per CDNA5 ISA 7.12.2.
// ---------------------------------------------------------------------------
#define BM 64
#define BN 128
#define BK 32
#define LDT 40   // padded LDS row length (halfs) to spread banks
#define KSTEPS (Dm / BK)

__global__ __launch_bounds__(256)
void k_gemm(const _Float16* __restrict__ ctx, const _Float16* __restrict__ w16,
            const float* __restrict__ bias, float* __restrict__ out) {
  int mb = blockIdx.x & 7;              // Sev/BM  = 8
  int nb = blockIdx.x >> 3;             // Dm/BN   = 8
  int b  = blockIdx.y;
  int row0 = mb * BM;
  int col0 = nb * BN;

  const _Float16* A = ctx + (size_t)b * Sev * Dm;

  __shared__ _Float16 As[2][BM * LDT];  //  5 KB x2
  __shared__ _Float16 Bs[2][BN * LDT];  // 10 KB x2

  int t    = threadIdx.x;
  int lane = t & 31;
  int wv   = t >> 5;                    // 0..7
  int mTile = wv & 3;                   // 4 m-tiles of 16 rows
  int nGrp  = wv >> 2;                  // 0..1, each wave owns 4 n-tiles

  bool hi = lane >= 16;
  int  lm = lane & 15;
  // A fragment K chunks: lanes 0-15 -> K 0..7 & 16..23 ; lanes 16-31 -> 8..15 & 24..31
  int aC0 = hi ? 8 : 0;
  int aC1 = aC0 + 16;
  // B fragment: lanes 0-15 -> K 0..7 & 16..23 ; lanes 16-31 -> 8..15 & 24..31
  int bC0 = hi ? 8 : 0;
  int aRow = mTile * 16 + lm;

  // cooperative staging: A 8 halfs/thread, B 16 halfs/thread
  int ar = t >> 2, ac = (t & 3) * 8;
  int br = t >> 1, bc = (t & 1) * 16;
  const _Float16* ga = A   + (size_t)(row0 + ar) * Dm + ac;
  const _Float16* gb = w16 + (size_t)(col0 + br) * Dm + bc;

  const v8f vzero = {0.f, 0.f, 0.f, 0.f, 0.f, 0.f, 0.f, 0.f};
  v8f acc[4];
#pragma unroll
  for (int nt = 0; nt < 4; nt++) acc[nt] = vzero;

  // prologue: stage tile 0
  v4f ra  = *reinterpret_cast<const v4f*>(ga);
  v4f rb0 = *reinterpret_cast<const v4f*>(gb);
  v4f rb1 = *reinterpret_cast<const v4f*>(gb + 8);
  *reinterpret_cast<v4f*>(&As[0][ar * LDT + ac])     = ra;
  *reinterpret_cast<v4f*>(&Bs[0][br * LDT + bc])     = rb0;
  *reinterpret_cast<v4f*>(&Bs[0][br * LDT + bc + 8]) = rb1;
  __syncthreads();

  for (int it = 0; it < KSTEPS; ++it) {
    int cur = it & 1;
    int nxt = cur ^ 1;
    bool more = (it + 1) < KSTEPS;

    // prefetch next tile global->reg (latency hidden under the WMMA chain)
    if (more) {
      int k0 = (it + 1) * BK;
      ra  = *reinterpret_cast<const v4f*>(ga + k0);
      rb0 = *reinterpret_cast<const v4f*>(gb + k0);
      rb1 = *reinterpret_cast<const v4f*>(gb + k0 + 8);
    }

    // preload all fragments, then issue 4 WMMAs back-to-back
    Frag16 ua;
    ua.f[0] = *reinterpret_cast<const v4f*>(&As[cur][aRow * LDT + aC0]);
    ua.f[1] = *reinterpret_cast<const v4f*>(&As[cur][aRow * LDT + aC1]);
    Frag16 ub[4];
#pragma unroll
    for (int nt = 0; nt < 4; nt++) {
      int bRow = nGrp * 64 + nt * 16 + lm;
      ub[nt].f[0] = *reinterpret_cast<const v4f*>(&Bs[cur][bRow * LDT + bC0]);
      ub[nt].f[1] = *reinterpret_cast<const v4f*>(&Bs[cur][bRow * LDT + bC0 + 16]);
    }
#pragma unroll
    for (int nt = 0; nt < 4; nt++) {
      acc[nt] = __builtin_amdgcn_wmma_f32_16x16x32_f16(
          false, ua.h, false, ub[nt].h, (short)0, acc[nt], false, false);
    }

    // store prefetched tile into the other buffer (no conflict with readers)
    if (more) {
      *reinterpret_cast<v4f*>(&As[nxt][ar * LDT + ac])     = ra;
      *reinterpret_cast<v4f*>(&Bs[nxt][br * LDT + bc])     = rb0;
      *reinterpret_cast<v4f*>(&Bs[nxt][br * LDT + bc + 8]) = rb1;
    }
    __syncthreads();  // single barrier per k-step
  }

  // epilogue: C/D layout -> VGPR r holds row r (lanes 0-15) / r+8 (lanes 16-31)
  int outRowBase = row0 + mTile * 16 + (hi ? 8 : 0);
  float* ob = out + (size_t)b * Sev * Dm;
#pragma unroll
  for (int nt = 0; nt < 4; nt++) {
    int col = col0 + nGrp * 64 + nt * 16 + lm;
    float bz = bias[col];
#pragma unroll
    for (int r = 0; r < 8; r++) {
      ob[(size_t)(outRowBase + r) * Dm + col] = acc[nt][r] + bz;
    }
  }
}

// ---------------------------------------------------------------------------
// Kernel 3: banded scores + 2-entry softmax.
// The mask keeps only k = q±1, so softmax reduces to two adjacent dots.
// One 256-thread block per (b,q).
// ---------------------------------------------------------------------------
__global__ __launch_bounds__(256)
void k_band(const float* __restrict__ qry, const float* __restrict__ key,
            const int* __restrict__ evm, float* __restrict__ Aup,
            float* __restrict__ Adn) {
  int row = blockIdx.x;                 // b*Sev + q
  int b = row / Sev, q = row % Sev;
  int t = threadIdx.x;

  bool vu = (q + 1 < Sev) && (evm[b * Sev + q + 1] != 0);
  bool vd = (q > 0)       && (evm[b * Sev + q - 1] != 0);

  const float* Q  = qry + (size_t)row * Dm;
  const float* Ku = key + (size_t)(vu ? row + 1 : row) * Dm;
  const float* Kd = key + (size_t)(vd ? row - 1 : row) * Dm;

  float su = 0.f, sd = 0.f;
#pragma unroll
  for (int i = 0; i < 4; i++) {
    int d = t + 256 * i;
    float qv = Q[d];
    su += qv * Ku[d];
    sd += qv * Kd[d];
  }
  for (int o = 16; o > 0; o >>= 1) {
    su += __shfl_xor(su, o, 32);
    sd += __shfl_xor(sd, o, 32);
  }
  __shared__ float s1[8], s2[8];
  int w = t >> 5, l = t & 31;
  if (l == 0) { s1[w] = su; s2[w] = sd; }
  __syncthreads();
  if (t == 0) {
    float fu = 0.f, fd = 0.f;
    for (int i = 0; i < 8; i++) { fu += s1[i]; fd += s2[i]; }
    const float scl = 2.0f / (float)Dm;    // 1/(D/2)
    float s_up = fu * scl, s_dn = fd * scl;
    float up, dn;
    if (vu && vd) {
      float m  = fmaxf(s_up, s_dn);
      float eu = __expf(s_up - m), ed = __expf(s_dn - m);
      float inv = 1.0f / (eu + ed);
      up = eu * inv; dn = ed * inv;
    } else if (vu) { up = 1.0f; dn = 0.0f; }
    else if (vd)   { up = 0.0f; dn = 1.0f; }
    else           { up = 1.0f / Sev; dn = 1.0f / Sev; }  // degenerate row
    Aup[row] = up;
    Adn[row] = dn;
  }
}

// ---------------------------------------------------------------------------
// Kernel 4a: A_new <- attention_mask
// ---------------------------------------------------------------------------
__global__ void k_anew_init(const float* __restrict__ am, float* __restrict__ Anew) {
  int i = blockIdx.x * blockDim.x + threadIdx.x;
  if (i < Bn * SFULL) Anew[i] = am[i];
}

// ---------------------------------------------------------------------------
// Kernel 4b: scatter A_d1[q] = sqrt(A[q,q+1]*A[q+1,q] + eps) to loc[b,q]
// ---------------------------------------------------------------------------
__global__ void k_scatter(const int* __restrict__ loc, const float* __restrict__ Aup,
                          const float* __restrict__ Adn, float* __restrict__ Anew) {
  int i = blockIdx.x * blockDim.x + threadIdx.x;  // b*(Sev-1) + q
  if (i >= Bn * (Sev - 1)) return;
  int b = i / (Sev - 1), q = i % (Sev - 1);
  float v = sqrtf(Aup[b * Sev + q] * Adn[b * Sev + q + 1] + EPSc);
  int j = loc[b * Sev + q];
  if (j < SFULL) Anew[b * SFULL + j] = v;
}

// ---------------------------------------------------------------------------
// Kernel 5: per-batch exclusive prefix scan of
//   L[j] = log(prior[j,j+1] + (1-prior[j,j+1])*Anew[j] + eps),  j < SFULL-1
// (tri@logA@tri telescopes into prefix sums of the superdiagonal logs)
// ---------------------------------------------------------------------------
__global__ __launch_bounds__(1024)
void k_scan(const float* __restrict__ Anew, const float* __restrict__ prior,
            float* __restrict__ P) {
  int b = blockIdx.x, j = threadIdx.x;
  __shared__ float s[SFULL];
  float L = 0.f;
  if (j < SFULL - 1) {
    float av = Anew[b * SFULL + j];
    float pr = prior[(size_t)b * SFULL * SFULL + (size_t)j * SFULL + j + 1];
    float v  = pr + (1.0f - pr) * av;
    L = logf(v + EPSc);
  }
  s[j] = L;
  __syncthreads();
  float val = L;
  for (int off = 1; off < SFULL; off <<= 1) {
    float add = (j >= off) ? s[j - off] : 0.f;
    __syncthreads();
    val += add;
    s[j] = val;
    __syncthreads();
  }
  P[b * SFULL + j] = val - L;           // exclusive prefix
}

// ---------------------------------------------------------------------------
// Kernel 6: fused output writer (memory-bound part, 16B vectorized)
//   C_prior[b,i,j] = exp(P[max(i,j)] - P[min(i,j)])   (diag -> exp(0) = 1)
//   A_new_full[b,i,j] = prior + (1-prior)*M,  M tridiagonal from A_new
// One 256-thread block per (b,i) row.
// ---------------------------------------------------------------------------
__global__ __launch_bounds__(256)
void k_out(const float* __restrict__ prior, const float* __restrict__ P,
           const float* __restrict__ Anew, float* __restrict__ Cout,
           float* __restrict__ Aout) {
  int bi = blockIdx.x;                  // b*SFULL + i
  int b = bi >> 10, i = bi & (SFULL - 1);
  const float* Prow = P + b * SFULL;
  float Pi  = Prow[i];
  float aup = (i < SFULL - 1) ? Anew[b * SFULL + i]     : 0.f;  // M[i,i+1]
  float adn = (i > 0)         ? Anew[b * SFULL + i - 1] : 0.f;  // M[i,i-1]

  size_t base = (size_t)bi * SFULL;
  int t = threadIdx.x;
  int j0 = t * 4;
  v4f pj = reinterpret_cast<const v4f*>(Prow)[t];
  v4f pr = reinterpret_cast<const v4f*>(prior + base)[t];

  v4f c, a;
#pragma unroll
  for (int e = 0; e < 4; e++) {
    int j = j0 + e;
    float pjv = pj[e];
    float d = (j > i) ? (pjv - Pi) : (Pi - pjv);
    c[e] = __expf(d);
    float m = (j == i + 1) ? aup : ((j == i - 1) ? adn : 0.f);
    float prv = pr[e];
    a[e] = prv + (1.0f - prv) * m;
  }
  reinterpret_cast<v4f*>(Cout + base)[t] = c;
  reinterpret_cast<v4f*>(Aout + base)[t] = a;
}

// ---------------------------------------------------------------------------
// Host launch
// ---------------------------------------------------------------------------
extern "C" void kernel_launch(void* const* d_in, const int* in_sizes, int n_in,
                              void* d_out, int out_size, void* d_ws, size_t ws_size,
                              hipStream_t stream) {
  (void)in_sizes; (void)n_in; (void)out_size; (void)ws_size;

  const float* hs    = (const float*)d_in[0];   // (B, SFULL, D)
  const float* am    = (const float*)d_in[1];   // (B, SFULL)
  const float* prior = (const float*)d_in[2];   // (B, SFULL, SFULL)
  const float* Wk    = (const float*)d_in[3];   // (D, D)
  const float* bk    = (const float*)d_in[4];   // (D,)
  const float* Wq    = (const float*)d_in[5];   // (D, D)
  const float* bq    = (const float*)d_in[6];   // (D,)
  const float* lnw   = (const float*)d_in[7];   // (D,)
  const float* lnb   = (const float*)d_in[8];   // (D,)
  const int*   loc   = (const int*)d_in[9];     // (B, Sev)
  const int*   evm   = (const int*)d_in[10];    // (B, Sev)

  float* out  = (float*)d_out;
  float* Cout = out;                                   // first output
  float* Aout = out + (size_t)Bn * SFULL * SFULL;      // second output

  // workspace layout
  char* ws = (char*)d_ws;
  _Float16* ctx16 = (_Float16*)ws;                        ws += (size_t)Bn * Sev * Dm * 2;
  _Float16* wk16  = (_Float16*)ws;                        ws += (size_t)Dm * Dm * 2;
  _Float16* wq16  = (_Float16*)ws;                        ws += (size_t)Dm * Dm * 2;
  float*    keyf  = (float*)ws;                           ws += (size_t)Bn * Sev * Dm * 4;
  float*    qryf  = (float*)ws;                           ws += (size_t)Bn * Sev * Dm * 4;
  float*    Aup   = (float*)ws;                           ws += (size_t)Bn * Sev * 4;
  float*    Adn   = (float*)ws;                           ws += (size_t)Bn * Sev * 4;
  float*    Anew  = (float*)ws;                           ws += (size_t)Bn * SFULL * 4;
  float*    P     = (float*)ws;                           ws += (size_t)Bn * SFULL * 4;

  // 0) weights -> f16
  k_cvt_w<<<(Dm * Dm + 255) / 256, 256, 0, stream>>>(Wk, Wq, wk16, wq16);
  // 1) gather + layernorm -> f16 ctx
  k_gather_ln<<<Bn * Sev, 256, 0, stream>>>(hs, loc, lnw, lnb, ctx16);
  // 2) WMMA GEMMs: key = ctx@Wk^T + bk, query = ctx@Wq^T + bq
  dim3 gg(64, Bn);
  k_gemm<<<gg, 256, 0, stream>>>(ctx16, wk16, bk, keyf);
  k_gemm<<<gg, 256, 0, stream>>>(ctx16, wq16, bq, qryf);
  // 3) banded scores + softmax
  k_band<<<Bn * Sev, 256, 0, stream>>>(qryf, keyf, evm, Aup, Adn);
  // 4) build A_new (init then scatter)
  k_anew_init<<<(Bn * SFULL + 255) / 256, 256, 0, stream>>>(am, Anew);
  k_scatter<<<(Bn * (Sev - 1) + 255) / 256, 256, 0, stream>>>(loc, Aup, Adn, Anew);
  // 5) per-batch prefix scan of superdiagonal logs
  k_scan<<<Bn, SFULL, 0, stream>>>(Anew, prior, P);
  // 6) fused output writer
  k_out<<<Bn * SFULL, 256, 0, stream>>>(prior, P, Anew, Cout, Aout);
}
